// LSTM_85452669321829
// MI455X (gfx1250) — matrix operand order repaired and checked by reading, not compiled, not run on hardware
//
#include <hip/hip_runtime.h>
#include <hip/hip_bf16.h>
#include <cstdint>
#include <cstddef>

// ---------------------------------------------------------------------------
// Persistent WMMA LSTM for MI455X (gfx1250, wave32).
//   - Weights repacked fp32 -> bf16 into per-lane WMMA A-fragment order (4 MB)
//   - Each of 32 persistent workgroups keeps its 128 KB weight slice in LDS
//     (staged via GLOBAL_LOAD_ASYNC_TO_LDS_B128 when the builtin exists)
//   - Per step: GEMV via v_wmma_f32_16x16x32_bf16 with 4 independent
//     accumulators (breaks the serial D->C chain), gate math on 16 channels,
//     2 KB h-exchange + device-wide generation barrier
// ---------------------------------------------------------------------------

#define T_STEPS 32768
#define XD 512
#define CD 512
#define ID 1024
#define NWG 32                       // persistent workgroups
#define CH_PER_WG 16                 // channels owned per WG
#define KCHUNKS (ID / 32)            // 32 K-chunks of 32
#define HALF_PER_WG (4 * KCHUNKS * 32 * 16)   // 65536 bf16 halves = 128 KB

typedef __bf16 v16bf __attribute__((ext_vector_type(16)));
typedef float  v8f   __attribute__((ext_vector_type(8)));

struct alignas(16) Bytes32 { uint4 lo, hi; };

static __device__ __forceinline__ unsigned short f2bf(float x) {
  unsigned u = __float_as_uint(x);
  u += 0x7FFFu + ((u >> 16) & 1u);          // round-to-nearest-even
  return (unsigned short)(u >> 16);
}

static __device__ __forceinline__ v16bf load16bf(const unsigned short* p) {
  Bytes32 t = *(const Bytes32*)p;           // two 16B LDS loads
  return __builtin_bit_cast(v16bf, t);
}

// ---------------------------------------------------------------------------
// Pack W rows (gate t, channel-block g) into this kernel's A-fragment lane order (contiguous k):
//   lane L holds row M = L%16;  half j -> K = (L&16) + j
// Flat layout: [g][gate][kchunk][lane][j]
// ---------------------------------------------------------------------------
__global__ void lstm_pack_w_85452669321829(const float* __restrict__ wf,
                                           const float* __restrict__ wi,
                                           const float* __restrict__ wc,
                                           const float* __restrict__ wo,
                                           unsigned short* __restrict__ wpack) {
  unsigned e = blockIdx.x * 256u + threadIdx.x;   // < NWG*HALF_PER_WG = 2M
  unsigned j     = e & 15u;
  unsigned L     = (e >> 4) & 31u;
  unsigned chunk = (e >> 9) & 31u;
  unsigned t     = (e >> 14) & 3u;
  unsigned g     = e >> 16;
  unsigned row   = g * CH_PER_WG + (L & 15u);
  unsigned K     = chunk * 32u + (L & 16u) + j;
  const float* W = (t == 0u) ? wf : (t == 1u) ? wi : (t == 2u) ? wc : wo;
  wpack[e] = f2bf(W[row * ID + K]);
}

__global__ void lstm_init_85452669321829(float* __restrict__ hbuf,
                                         unsigned int* __restrict__ bar) {
  int i = threadIdx.x;
  for (int e = i; e < CD; e += 256) hbuf[e] = 0.0f;
  if (i == 0) *bar = 0u;
}

// ---------------------------------------------------------------------------
// Persistent kernel: 32 WGs x 128 threads (4 waves). Wave w handles gate w.
// ---------------------------------------------------------------------------
__launch_bounds__(128, 1)
__global__ void lstm_persist_85452669321829(const float* __restrict__ x,
                                            const float* __restrict__ bfv,
                                            const float* __restrict__ biv,
                                            const float* __restrict__ bcv,
                                            const float* __restrict__ bov,
                                            const unsigned short* __restrict__ wpack,
                                            float* __restrict__ hbuf,
                                            unsigned int* __restrict__ bar,
                                            float* __restrict__ out) {
  extern __shared__ char smem[];
  unsigned short* lds_w  = (unsigned short*)smem;              // 131072 B
  unsigned short* lds_iv = (unsigned short*)(smem + 131072);   // 2048 B
  float*          lds_g  = (float*)(smem + 131072 + 2048);     // 256 B

  const int g    = blockIdx.x;
  const int tid  = threadIdx.x;
  const int wave = tid >> 5;
  const int lane = tid & 31;

  // ---- Stage this WG's 128 KB bf16 weight slice into LDS once. ----
#if __has_builtin(__builtin_amdgcn_global_load_async_to_lds_b128) && \
    __has_builtin(__builtin_amdgcn_s_wait_asynccnt)
  {
    // gfx1250 async DMA path: no VGPR round-trip, tracked by ASYNCcnt.
    // Builtin params (from hipcc diagnostic): v4i AS(1)* gsrc, v4i AS(3)* ldst,
    // imm offset, imm cpol.
    typedef __attribute__((__vector_size__(4 * sizeof(int)))) int v4i_t;
    typedef __attribute__((address_space(1))) v4i_t* gv4_t;
    typedef __attribute__((address_space(3))) v4i_t* lv4_t;
    const char* gsrc = (const char*)(wpack + (size_t)g * HALF_PER_WG);
    char* ldst = (char*)lds_w;
    for (int e = tid; e < HALF_PER_WG / 8; e += 128) {
      __builtin_amdgcn_global_load_async_to_lds_b128(
          (gv4_t)(gsrc + (size_t)e * 16), (lv4_t)(ldst + (size_t)e * 16),
          0, 0);
    }
    __builtin_amdgcn_s_wait_asynccnt(0);
  }
#else
  {
    const uint4* wsrc = (const uint4*)(wpack + (size_t)g * HALF_PER_WG);
    uint4* wdst = (uint4*)lds_w;
    for (int e = tid; e < HALF_PER_WG / 8; e += 128) wdst[e] = wsrc[e];
  }
#endif

  const int ch = g * CH_PER_WG + tid;       // valid for tid < 16
  float bF = 0.f, bI = 0.f, bC = 0.f, bO = 0.f;
  float cval = 0.f, hval = 0.f;
  if (tid < 16) { bF = bfv[ch]; bI = biv[ch]; bC = bcv[ch]; bO = bov[ch]; }

  __syncthreads();

  for (int t = 0; t < T_STEPS; ++t) {
    // Build bf16 input vector [x_t ; h_{t-1}] in LDS.
    for (int e = tid; e < ID; e += 128) {
      float v = (e < XD) ? x[(size_t)t * XD + e]
                         : ((t == 0) ? 0.f : hbuf[e - XD]);
      lds_iv[e] = f2bf(v);
    }
    __syncthreads();

    // Prefetch next timestep's x row (global_prefetch_b8), 16 lines cover 2KB.
    if (t + 1 < T_STEPS && tid < 16)
      __builtin_prefetch(&x[(size_t)(t + 1) * XD + tid * 32], 0, 1);

    // Wave 'wave' computes its gate's 16 rows. Four independent accumulators
    // break the serial WMMA D->C dependency chain (32 -> 8 deep).
    v8f acc0 = {}, acc1 = {}, acc2 = {}, acc3 = {};
    const int bsel = (lane & 16);           // B frag: K 0..15 vs 16..31
#pragma unroll
    for (int kc = 0; kc < KCHUNKS; kc += 4) {
      v16bf a0 = load16bf(lds_w + (((wave * KCHUNKS + kc + 0) * 32 + lane) << 4));
      v16bf b0 = load16bf(lds_iv + (kc + 0) * 32 + bsel);
      acc0 = __builtin_amdgcn_wmma_f32_16x16x32_bf16(
          false, a0, false, b0, (short)0, acc0, false, false);
      v16bf a1 = load16bf(lds_w + (((wave * KCHUNKS + kc + 1) * 32 + lane) << 4));
      v16bf b1 = load16bf(lds_iv + (kc + 1) * 32 + bsel);
      acc1 = __builtin_amdgcn_wmma_f32_16x16x32_bf16(
          false, a1, false, b1, (short)0, acc1, false, false);
      v16bf a2 = load16bf(lds_w + (((wave * KCHUNKS + kc + 2) * 32 + lane) << 4));
      v16bf b2 = load16bf(lds_iv + (kc + 2) * 32 + bsel);
      acc2 = __builtin_amdgcn_wmma_f32_16x16x32_bf16(
          false, a2, false, b2, (short)0, acc2, false, false);
      v16bf a3 = load16bf(lds_w + (((wave * KCHUNKS + kc + 3) * 32 + lane) << 4));
      v16bf b3 = load16bf(lds_iv + (kc + 3) * 32 + bsel);
      acc3 = __builtin_amdgcn_wmma_f32_16x16x32_bf16(
          false, a3, false, b3, (short)0, acc3, false, false);
    }
    v8f acc = (acc0 + acc1) + (acc2 + acc3);

    // D layout: lanes 0-15 hold M=0..7 in v0..v7, lanes 16-31 hold M=8..15.
    if (lane == 0) {
#pragma unroll
      for (int m = 0; m < 8; ++m) lds_g[wave * 16 + m] = acc[m];
    } else if (lane == 16) {
#pragma unroll
      for (int m = 0; m < 8; ++m) lds_g[wave * 16 + 8 + m] = acc[m];
    }
    __syncthreads();

    // Gate math + state update on this WG's 16 channels.
    if (tid < 16) {
      float gf = lds_g[ 0 + tid] + bF;
      float gi = lds_g[16 + tid] + bI;
      float gc = lds_g[32 + tid] + bC;
      float go = lds_g[48 + tid] + bO;
      float f  = 1.f / (1.f + __expf(-gf));
      float i  = 1.f / (1.f + __expf(-gi));
      float cc = tanhf(gc);
      float o  = 1.f / (1.f + __expf(-go));
      cval = f * cval + i * cc;
      hval = tanhf(cval) * o;
      hbuf[ch] = hval;
    }
    __threadfence();
    __syncthreads();

    // Device-wide arrive/spin barrier (monotonic generation counter).
    if (tid == 0) {
      atomicAdd(bar, 1u);
      const unsigned int target = (unsigned int)(t + 1) * NWG;
      while (__hip_atomic_load(bar, __ATOMIC_ACQUIRE,
                               __HIP_MEMORY_SCOPE_AGENT) < target)
        __builtin_amdgcn_s_sleep(1);
    }
    __syncthreads();
    __threadfence();   // acquire: next iteration's hbuf reads see fresh data
  }

  if (tid < 16) { out[ch] = cval; out[CD + ch] = hval; }
}

extern "C" void kernel_launch(void* const* d_in, const int* in_sizes, int n_in,
                              void* d_out, int out_size, void* d_ws, size_t ws_size,
                              hipStream_t stream) {
  const float* x   = (const float*)d_in[0];
  const float* wfw = (const float*)d_in[1];
  const float* wfb = (const float*)d_in[2];
  const float* wiw = (const float*)d_in[3];
  const float* wib = (const float*)d_in[4];
  const float* wcw = (const float*)d_in[5];
  const float* wcb = (const float*)d_in[6];
  const float* wow = (const float*)d_in[7];
  const float* wob = (const float*)d_in[8];
  float* out = (float*)d_out;

  char* ws = (char*)d_ws;
  unsigned short* wpack = (unsigned short*)ws;  // 4 MiB packed bf16 weights
  const size_t wbytes = (size_t)NWG * HALF_PER_WG * sizeof(unsigned short);
  float* hbuf = (float*)(ws + wbytes);          // 2 KiB h exchange buffer
  unsigned int* bar = (unsigned int*)(ws + wbytes + 4096);

  lstm_pack_w_85452669321829<<<(NWG * HALF_PER_WG) / 256, 256, 0, stream>>>(
      wfw, wiw, wcw, wow, wpack);
  lstm_init_85452669321829<<<1, 256, 0, stream>>>(hbuf, bar);

  const int smem = 131072 + 2048 + 256;   // 130.25 KB dynamic LDS
  (void)hipFuncSetAttribute((const void*)lstm_persist_85452669321829,
                            hipFuncAttributeMaxDynamicSharedMemorySize, smem);
  lstm_persist_85452669321829<<<NWG, 128, smem, stream>>>(
      x, wfb, wib, wcb, wob, wpack, hbuf, bar, out);
}
